// CRATE_26534307955025
// MI455X (gfx1250) — compile-verified, matrix operand order checked by reading, not run on hardware
//
#include <hip/hip_runtime.h>

// ---------------------------------------------------------------------------
// MI455X (gfx1250, CDNA5) transformer forward.
//   wave32, WMMA 16x16x32 bf16 -> f32 accumulate.
//   Memory-bound model: without fusion, materialized attention scores cost
//   ~3.6 GB of HBM traffic (vs 0.6 GB of weights), so attention is computed
//   flash-style in one kernel (scores live in registers/LDS). f32 weights are
//   streamed once and converted to bf16 in-register; activations kept bf16 and
//   staged with GLOBAL_LOAD_ASYNC_TO_LDS_B128 (ASYNCcnt path).
// ---------------------------------------------------------------------------

typedef __bf16 bf16;
typedef __attribute__((ext_vector_type(8)))  bf16  v8bf;
typedef __attribute__((ext_vector_type(16))) bf16  v16bf;
typedef __attribute__((ext_vector_type(8)))  float v8f;

#define B_  4
#define T_  1024
#define E_  768
#define L_  12
#define H_  6
#define HD_ 128
#define HID_ 3072
#define V_  50304
#define EPS_ 1e-6f
#define NEGINF_ -3.0e38f

// round-to-nearest-even f32 -> bf16
__device__ __forceinline__ bf16 f2bf(float f) {
  union { float f; unsigned int u; } v; v.f = f;
  unsigned int r = v.u + 0x7FFFu + ((v.u >> 16) & 1u);
  unsigned short h = (unsigned short)(r >> 16);
  bf16 out;
  __builtin_memcpy(&out, &h, 2);
  return out;
}

__device__ __forceinline__ v8f wmma_bf16(v16bf a, v16bf b, v8f c) {
  return __builtin_amdgcn_wmma_f32_16x16x32_bf16(
      false, a, false, b, (short)0, c, false, false);
}

// Async copy 16B/lane global -> LDS (GLOBAL_LOAD_ASYNC_TO_LDS_B128, ASYNCcnt).
// LDS address = low 32 bits of the generic pointer (CDNA5 LDS aperture rule).
__device__ __forceinline__ void async_ld_b128(const bf16* gptr, bf16* lptr) {
  unsigned lds = (unsigned)(unsigned long long)lptr;
  asm volatile("global_load_async_to_lds_b128 %0, %1, off"
               :: "v"(lds), "v"(gptr) : "memory");
}
__device__ __forceinline__ void wait_async0() {
  asm volatile("s_wait_asynccnt 0x0" ::: "memory");
}

// v16bf WMMA fragment: elements 0..7 = K[k0..k0+7], 8..15 = K[k0+16..k0+23]
__device__ __forceinline__ v16bf ld_frag(const bf16* p, int k0) {
  v8bf lo = *(const v8bf*)(p + k0);
  v8bf hi = *(const v8bf*)(p + k0 + 16);
  v16bf f;
#pragma unroll
  for (int i = 0; i < 8; ++i) { f[i] = lo[i]; f[i + 8] = hi[i]; }
  return f;
}

__device__ __forceinline__ float block_red(float v, float* buf, int tid, int n, int domax) {
  buf[tid] = v;
  __syncthreads();
  for (int s = n >> 1; s > 0; s >>= 1) {
    if (tid < s) buf[tid] = domax ? fmaxf(buf[tid], buf[tid + s]) : (buf[tid] + buf[tid + s]);
    __syncthreads();
  }
  float r = buf[0];
  __syncthreads();
  return r;
}

// ---------------------------------------------------------------------------
// Generic WMMA GEMM:  C[z][M,N] = A[z][M,K] * W[z][N,K]^T   (same as round 2)
// ---------------------------------------------------------------------------
__global__ __launch_bounds__(256)
void crate_gemm_k(const bf16* __restrict__ A,
                  const float* __restrict__ Wf, const bf16* __restrict__ Wb,
                  float* Cf, bf16* Cb, const float* __restrict__ thr,
                  int K, int lda, int ldw, int ldc,
                  long long aZ, long long wZ, long long cZo, long long cZi, int zdiv,
                  float scale, int epi) {
  __shared__ bf16 lsA[64 * 72];
  __shared__ bf16 lsW[128 * 72];

  const int tid = threadIdx.x;
  const int z = blockIdx.z;
  const int nBase = blockIdx.x * 128;
  const int mBase = blockIdx.y * 64;

  const bf16* Az = A + (long long)z * aZ;
  const float* Wfz = Wf ? Wf + (long long)z * wZ : nullptr;
  const bf16* Wbz = Wb ? Wb + (long long)z * wZ : nullptr;

  const int ar = tid >> 2;            // 0..63   A row, 2x16B per thread
  const int ak = (tid & 3) * 16;
  const int wr = tid >> 1;            // 0..127  W row, 32 elems per thread
  const int wk = (tid & 1) * 32;

  const int wv = tid >> 5;
  const int wMB = (wv >> 2) * 32;
  const int wNB = (wv & 3) * 32;
  const int lane = tid & 31;
  const int ln = lane & 15;
  const int kh = lane >> 4;

  v8f acc00 = {}, acc01 = {}, acc10 = {}, acc11 = {};

  for (int k0 = 0; k0 < K; k0 += 64) {
    {
      const bf16* gA = Az + (long long)(mBase + ar) * lda + k0 + ak;
      async_ld_b128(gA,     &lsA[ar * 72 + ak]);
      async_ld_b128(gA + 8, &lsA[ar * 72 + ak + 8]);
    }
    if (Wbz) {
      const bf16* g = Wbz + (long long)(nBase + wr) * ldw + k0 + wk;
#pragma unroll
      for (int j = 0; j < 4; ++j)
        async_ld_b128(g + j * 8, &lsW[wr * 72 + wk + j * 8]);
    } else {
      const float* g = Wfz + (long long)(nBase + wr) * ldw + k0 + wk;
      if (k0 + 64 < K) __builtin_prefetch(g + 64, 0, 1);
#pragma unroll
      for (int j = 0; j < 8; ++j) {
        float4 f = *(const float4*)(g + j * 4);
        bf16* d = &lsW[wr * 72 + wk + j * 4];
        d[0] = f2bf(f.x); d[1] = f2bf(f.y); d[2] = f2bf(f.z); d[3] = f2bf(f.w);
      }
    }
    wait_async0();
    __syncthreads();

#pragma unroll
    for (int kk = 0; kk < 2; ++kk) {
      const int kf = kk * 32 + kh * 8;
      v16bf a0 = ld_frag(&lsA[(wMB + ln) * 72], kf);
      v16bf a1 = ld_frag(&lsA[(wMB + 16 + ln) * 72], kf);
      v16bf b0 = ld_frag(&lsW[(wNB + ln) * 72], kf);
      v16bf b1 = ld_frag(&lsW[(wNB + 16 + ln) * 72], kf);
      acc00 = wmma_bf16(a0, b0, acc00);
      acc01 = wmma_bf16(a0, b1, acc01);
      acc10 = wmma_bf16(a1, b0, acc10);
      acc11 = wmma_bf16(a1, b1, acc11);
    }
    __syncthreads();
  }

  const long long cOff = (long long)(z / zdiv) * cZo + (long long)(z % zdiv) * cZi;
#pragma unroll
  for (int tm = 0; tm < 2; ++tm) {
#pragma unroll
    for (int tn = 0; tn < 2; ++tn) {
      v8f acc = tm ? (tn ? acc11 : acc10) : (tn ? acc01 : acc00);
      const int col = nBase + wNB + tn * 16 + ln;
#pragma unroll
      for (int r = 0; r < 8; ++r) {
        const int row = mBase + wMB + tm * 16 + kh * 8 + r;
        const float val = acc[r];
        const long long ci = cOff + (long long)row * ldc + col;
        if (epi == 0)      Cf[ci] = val * scale;
        else if (epi == 1) Cb[ci] = f2bf(val);
        else if (epi == 2) Cf[ci] += val;
        else               Cb[ci] = f2bf(fmaxf(val - thr[col], 0.f));
      }
    }
  }
}

// ---------------------------------------------------------------------------
// Fused flash attention (causal, q==k==v==w), one block per (b,h,128 q-rows).
//   8 waves; wave w owns q-rows [qb+16w, qb+16w+16).
//   Q: 4 v16bf A-fragments per wave (HD=128 = 4 K-chunks of 32), held in regs.
//   O: 16x128 f32 accumulator per wave (8 WMMA tiles).
//   K-loop: 64 keys per stage (async-to-LDS). Per 32-key pair:
//     8 WMMA (S=Q*K^T) -> scale+causal mask -> online softmax update
//     (row reductions via __shfl_xor over the 16-lane row groups)
//     P (bf16) -> wave-private LDS -> reload as A-fragment -> 8 WMMA (O+=P*V)
//   V comes from wT (d-major), so its B-fragments are K-contiguous.
// ---------------------------------------------------------------------------
#define PKT_ 136   // lsK pitch (64 keys x 128 d)
#define PVT_ 72    // lsVT pitch (128 d x 64 k)
__global__ __launch_bounds__(256)
void crate_fattn_k(const bf16* __restrict__ wqg, const bf16* __restrict__ wTg,
                   bf16* __restrict__ ob, float rscale) {
  __shared__ bf16 lsK[64 * PKT_];
  __shared__ bf16 lsVT[128 * PVT_];
  __shared__ bf16 lsP[8 * 16 * 32];   // per-wave P transpose patch

  const int tid = threadIdx.x;
  const int qi = blockIdx.x;              // q-block (128 rows)
  const int zz = blockIdx.y;              // b*H + h
  const int b = zz / H_, h = zz % H_;
  const int qb = qi * 128;

  const bf16* wq_bh = wqg + ((long long)zz * T_) * HD_;   // [T,HD]
  const bf16* wT_bh = wTg + ((long long)zz * HD_) * T_;   // [HD,T]

  const int wv = tid >> 5;
  const int lane = tid & 31;
  const int ln = lane & 15;
  const int kh = lane >> 4;

  // staging assignments
  const int kr = tid >> 2, kc = (tid & 3) * 32;   // lsK: 64 rows x 128
  const int vr = tid >> 1, vc = (tid & 1) * 32;   // lsVT: 128 rows x 64

  // ---- Q fragments (held in registers for the whole kernel) ----
  const bf16* qrow = wq_bh + (long long)(qb + wv * 16 + ln) * HD_;
  v16bf qfrag[4];
#pragma unroll
  for (int c = 0; c < 4; ++c) qfrag[c] = ld_frag(qrow, c * 32 + kh * 8);

  v8f oacc[8];
#pragma unroll
  for (int t = 0; t < 8; ++t) oacc[t] = (v8f){};
  float m[8], l[8];
#pragma unroll
  for (int r = 0; r < 8; ++r) { m[r] = NEGINF_; l[r] = 0.f; }

  bf16* myP = &lsP[wv * 16 * 32];
  const int nkb = 2 * qi + 2;             // causal: keys up to qb+127

  for (int kbi = 0; kbi < nkb; ++kbi) {
    const int kb = kbi * 64;
    // ---- stage K tile (64 keys x 128 d) and V^T tile (128 d x 64 k) ----
    {
      const bf16* gK = wq_bh + (long long)(kb + kr) * HD_ + kc;
#pragma unroll
      for (int j = 0; j < 4; ++j)
        async_ld_b128(gK + j * 8, &lsK[kr * PKT_ + kc + j * 8]);
      const bf16* gV = wT_bh + (long long)vr * T_ + kb + vc;
#pragma unroll
      for (int j = 0; j < 4; ++j)
        async_ld_b128(gV + j * 8, &lsVT[vr * PVT_ + vc + j * 8]);
    }
    wait_async0();
    __syncthreads();

#pragma unroll
    for (int pair = 0; pair < 2; ++pair) {
      // ---- S = Q * K^T for 32 keys (two 16x16 tiles) ----
      v8f s0 = {}, s1 = {};
#pragma unroll
      for (int c = 0; c < 4; ++c) {
        const int kf = c * 32 + kh * 8;
        s0 = wmma_bf16(qfrag[c], ld_frag(&lsK[(pair * 32 + ln) * PKT_], kf), s0);
        s1 = wmma_bf16(qfrag[c], ld_frag(&lsK[(pair * 32 + 16 + ln) * PKT_], kf), s1);
      }
      // ---- scale + causal mask + online softmax ----
      const int k0idx = kb + pair * 32 + ln;
      const int k1idx = k0idx + 16;
      float corr[8];
#pragma unroll
      for (int r = 0; r < 8; ++r) {
        const int q = qb + wv * 16 + kh * 8 + r;
        float a = (k0idx > q) ? NEGINF_ : s0[r] * rscale;
        float bb = (k1idx > q) ? NEGINF_ : s1[r] * rscale;
        float tmax = fmaxf(a, bb);
#pragma unroll
        for (int off = 8; off > 0; off >>= 1)
          tmax = fmaxf(tmax, __shfl_xor(tmax, off, 16));
        const float mnew = fmaxf(m[r], tmax);
        corr[r] = __expf(m[r] - mnew);
        m[r] = mnew;
        const float pa = __expf(a - mnew);
        const float pb = __expf(bb - mnew);
        float ts = pa + pb;
#pragma unroll
        for (int off = 8; off > 0; off >>= 1)
          ts += __shfl_xor(ts, off, 16);
        l[r] = l[r] * corr[r] + ts;
        myP[(kh * 8 + r) * 32 + ln] = f2bf(pa);
        myP[(kh * 8 + r) * 32 + 16 + ln] = f2bf(pb);
      }
#pragma unroll
      for (int t = 0; t < 8; ++t)
#pragma unroll
        for (int r = 0; r < 8; ++r) oacc[t][r] *= corr[r];
      // ---- O += P * V  (P reloaded as A-fragment; V from wT tile) ----
      v16bf pfrag = ld_frag(&myP[ln * 32], kh * 8);
#pragma unroll
      for (int t = 0; t < 8; ++t)
        oacc[t] = wmma_bf16(pfrag,
                            ld_frag(&lsVT[(t * 16 + ln) * PVT_], pair * 32 + kh * 8),
                            oacc[t]);
    }
    __syncthreads();
  }

  // ---- normalize and store O: ob[b, q, h*HD + d] ----
  float inv[8];
#pragma unroll
  for (int r = 0; r < 8; ++r) inv[r] = 1.f / l[r];
#pragma unroll
  for (int t = 0; t < 8; ++t) {
    const int col = h * HD_ + t * 16 + ln;
#pragma unroll
    for (int r = 0; r < 8; ++r) {
      const int q = qb + wv * 16 + kh * 8 + r;
      ob[((long long)b * T_ + q) * E_ + col] = f2bf(oacc[t][r] * inv[r]);
    }
  }
}

// ---- embedding gather + RMSNorm -> x, x0 (f32) ----
__global__ __launch_bounds__(256)
void crate_embed_k(const int* __restrict__ idx, const float* __restrict__ wte,
                   float* __restrict__ x, float* __restrict__ x0) {
  __shared__ float buf[256];
  const int bt = blockIdx.x, tid = threadIdx.x;
  const float* row = wte + (long long)idx[bt] * E_;
  float v[3]; float s = 0.f;
#pragma unroll
  for (int j = 0; j < 3; ++j) { v[j] = row[tid + j * 256]; s += v[j] * v[j]; }
  s = block_red(s, buf, tid, 256, 0);
  const float rn = rsqrtf(s / (float)E_ + EPS_);
#pragma unroll
  for (int j = 0; j < 3; ++j) {
    const long long o = (long long)bt * E_ + tid + j * 256;
    const float y = v[j] * rn;
    x[o] = y; x0[o] = y;
  }
}

// ---- x = rl*x + xl*x0 (optional), h = bf16(rmsnorm(x)) ----
__global__ __launch_bounds__(256)
void crate_residnorm_k(float* __restrict__ x, const float* __restrict__ x0,
                       const float* __restrict__ rlam, const float* __restrict__ xlam,
                       int layer, bf16* __restrict__ hout, int use_lambda) {
  __shared__ float buf[256];
  const int bt = blockIdx.x, tid = threadIdx.x;
  float* xr = x + (long long)bt * E_;
  const float* x0r = x0 + (long long)bt * E_;
  const float rl = use_lambda ? rlam[layer] : 1.f;
  const float xl = use_lambda ? xlam[layer] : 0.f;
  float v[3]; float s = 0.f;
#pragma unroll
  for (int j = 0; j < 3; ++j) {
    const int i = tid + j * 256;
    const float t = rl * xr[i] + xl * x0r[i];
    v[j] = t; s += t * t;
  }
  s = block_red(s, buf, tid, 256, 0);
  const float rn = rsqrtf(s / (float)E_ + EPS_);
#pragma unroll
  for (int j = 0; j < 3; ++j) {
    const int i = tid + j * 256;
    if (use_lambda) xr[i] = v[j];
    hout[(long long)bt * E_ + i] = f2bf(v[j] * rn);
  }
}

// ---- RoPE + per-head RMSNorm; emits w [B,H,T,HD] and wT [B,H,HD,T] bf16 ----
__global__ __launch_bounds__(64)
void crate_rope_k(const float* __restrict__ tmp, bf16* __restrict__ w,
                  bf16* __restrict__ wT) {
  __shared__ float buf[64];
  const int gid = blockIdx.x;        // ((b*T + t)*H + h)
  const int h = gid % H_;
  const int bt = gid / H_;
  const int t = bt % T_;
  const int b = bt / T_;
  const int i = threadIdx.x;         // 0..63 (pairs i, i+64)
  const float* in = tmp + (long long)bt * E_ + h * HD_;
  const float x1 = in[i], x2 = in[i + 64];
  const float ang = (float)t * __powf(10000.f, -(float)i / 64.f);
  float sn, cs;
  __sincosf(ang, &sn, &cs);
  const float o1 = x1 * cs + x2 * sn;
  const float o2 = -x1 * sn + x2 * cs;
  float ss = block_red(o1 * o1 + o2 * o2, buf, i, 64, 0);
  const float rn = rsqrtf(ss / (float)HD_ + EPS_);
  const long long base = ((long long)(b * H_ + h) * T_ + t) * HD_;
  w[base + i] = f2bf(o1 * rn);
  w[base + 64 + i] = f2bf(o2 * rn);
  const long long tbase = (long long)(b * H_ + h) * HD_ * T_;
  wT[tbase + (long long)i * T_ + t] = f2bf(o1 * rn);
  wT[tbase + (long long)(i + 64) * T_ + t] = f2bf(o2 * rn);
}

// ---------------------------------------------------------------------------
static void launch_gemm(hipStream_t stream, const bf16* A, const float* Wf, const bf16* Wb,
                        float* Cf, bf16* Cb, const float* thr,
                        int M, int N, int K, int lda, int ldw, int ldc,
                        long long aZ, long long wZ, long long cZo, long long cZi, int zdiv,
                        int Z, float scale, int epi) {
  dim3 g(N / 128, M / 64, Z), b(256);
  crate_gemm_k<<<g, b, 0, stream>>>(A, Wf, Wb, Cf, Cb, thr, K, lda, ldw, ldc,
                                    aZ, wZ, cZo, cZi, zdiv, scale, epi);
}

extern "C" void kernel_launch(void* const* d_in, const int* in_sizes, int n_in,
                              void* d_out, int out_size, void* d_ws, size_t ws_size,
                              hipStream_t stream) {
  const int*   idx    = (const int*)d_in[0];
  const float* wte    = (const float*)d_in[1];
  const float* lmw    = (const float*)d_in[2];
  const float* qkvw   = (const float*)d_in[3];
  const float* cprojw = (const float*)d_in[4];
  const float* dencw  = (const float*)d_in[5];
  const float* ddecw  = (const float*)d_in[6];
  const float* thr    = (const float*)d_in[7];
  const float* rlam   = (const float*)d_in[8];
  const float* xlam   = (const float*)d_in[9];
  float* out = (float*)d_out;

  const long long BT = (long long)B_ * T_;            // 4096

  // workspace carve-up (every byte fully rewritten each call)
  float* x    = (float*)d_ws;            // BT*E f32
  float* x0   = x + BT * E_;             // BT*E f32
  float* tmp  = x0 + BT * E_;            // BT*E f32 (qkv projection out)
  bf16* h    = (bf16*)(tmp + BT * E_);   // BT*E
  bf16* wq   = h + BT * E_;              // [B,H,T,HD]
  bf16* wT   = wq + BT * E_;             // [B,H,HD,T]
  bf16* ob   = wT + BT * E_;             // attention output [B,T,E]
  bf16* ab   = ob + BT * E_;             // ffn activation [B,T,HID]

  const float rscale = 0.08838834764831845f;  // HD^-0.5

  crate_embed_k<<<dim3((unsigned)BT), 256, 0, stream>>>(idx, wte, x, x0);

  for (int i = 0; i < L_; ++i) {
    // x = rl*x + xl*x0 ; h = rmsnorm(x)
    crate_residnorm_k<<<dim3((unsigned)BT), 256, 0, stream>>>(x, x0, rlam, xlam, i, h, 1);
    // qkv projection: tmp = h * qkv_w^T
    launch_gemm(stream, h, qkvw + (long long)i * E_ * E_, nullptr, tmp, nullptr, nullptr,
                (int)BT, E_, E_, E_, E_, E_, 0, 0, 0, 0, 1, 1, 1.f, 0);
    // RoPE + head rmsnorm -> w, wT (bf16)
    crate_rope_k<<<dim3((unsigned)(BT * H_)), 64, 0, stream>>>(tmp, wq, wT);
    // fused causal flash attention -> ob
    crate_fattn_k<<<dim3(T_ / 128, B_ * H_), 256, 0, stream>>>(wq, wT, ob, rscale);
    // x += o * cproj_w^T
    launch_gemm(stream, ob, cprojw + (long long)i * E_ * E_, nullptr, x, nullptr, nullptr,
                (int)BT, E_, E_, E_, E_, E_, 0, 0, 0, 0, 1, 1, 1.f, 2);
    // h2 = rmsnorm(x)
    crate_residnorm_k<<<dim3((unsigned)BT), 256, 0, stream>>>(x, x0, nullptr, nullptr, 0, h, 0);
    // a = relu(h2 * denc_w^T - thr)
    launch_gemm(stream, h, dencw + (long long)i * HID_ * E_, nullptr, nullptr, ab,
                thr + (long long)i * HID_,
                (int)BT, HID_, E_, E_, E_, HID_, 0, 0, 0, 0, 1, 1, 1.f, 3);
    // x += a * ddec_w^T
    launch_gemm(stream, ab, ddecw + (long long)i * E_ * HID_, nullptr, x, nullptr, nullptr,
                (int)BT, E_, HID_, HID_, HID_, E_, 0, 0, 0, 0, 1, 1, 1.f, 2);
  }

  // final norm + LM head
  crate_residnorm_k<<<dim3((unsigned)BT), 256, 0, stream>>>(x, x0, nullptr, nullptr, 0, h, 0);
  launch_gemm(stream, h, lmw, nullptr, out, nullptr, nullptr,
              (int)BT, V_, E_, E_, E_, V_, 0, 0, 0, 0, 1, 1, 1.f, 0);
}
